// SwinTransformerBlock_65678639891269
// MI455X (gfx1250) — matrix operand order repaired and checked by reading, not compiled, not run on hardware
//
#include <hip/hip_runtime.h>
#include <hip/hip_bf16.h>

typedef __attribute__((ext_vector_type(16))) _Float16 v16h;
typedef __attribute__((ext_vector_type(8)))  float    v8f;

#define HH     180
#define WW_    360
#define CC     192
#define WSZ    6
#define SHIFTV 3
#define NHEAD  6
#define MLON   60
#define NWIN   1800
#define NTOK   36
#define HDIM   32
#define NPAD   48
#define TOKENS (HH * WW_)            // 64800
#define SCALEV 0.17677669529663687f  // 32^-0.5

// ---------------- WMMA fragment helpers (CDNA5 16x16x32 f16 layouts) --------

__device__ __forceinline__ v16h load_a_frag(const _Float16* A, int lda,
                                            int row0, int k0, int lane) {
  // A is MxK row-major. lanes 0-15: rows M=0..15, K 0-15;
  // lanes 16-31: same rows, K 16-31. 16 contiguous halfs per lane.
  v16h a;
  const _Float16* p = A + (size_t)(row0 + (lane & 15)) * lda + k0 + ((lane >> 4) << 4);
#pragma unroll
  for (int i = 0; i < 16; ++i) a[i] = p[i];
  return a;
}

__device__ __forceinline__ v16h load_bt_frag(const _Float16* M, int ldm,
                                             int col0, int k0, int lane) {
  // B operand sourced from an NxK row-major matrix (transpose of logical KxN B).
  // 16 contiguous halfs per lane (two b128 loads).
  v16h b;
  const _Float16* p = M + (size_t)(col0 + (lane & 15)) * ldm + k0 + ((lane >> 4) << 4);
#pragma unroll
  for (int i = 0; i < 16; ++i) b[i] = p[i];
  return b;
}

__device__ __forceinline__ v16h load_b_frag(const _Float16* Bm, int ldb,
                                            int k0, int col0, int lane) {
  // Row-major KxN B operand (used only for LDS-resident V in attention).
  v16h b;
  const _Float16* p = Bm + (size_t)(k0 + ((lane >> 4) << 4)) * ldb + col0 + (lane & 15);
#pragma unroll
  for (int i = 0; i < 16; ++i) b[i] = p[(size_t)i * ldb];
  return b;
}

__device__ __forceinline__ v8f wmma16(v16h a, v16h b, v8f c) {
  return __builtin_amdgcn_wmma_f32_16x16x32_f16(false, a, false, b, (short)0, c,
                                                false, false);
}

// Software-pipelined k-loop: fragments for step t+1 are issued before the
// WMMAs of step t consume the current registers, so loads overlap matrix ops.
// X is (M,K) row-major with lda==K; WT is (N,K) row-major with ldw==K.
template <int K>
__device__ __forceinline__ void gemm_core(const _Float16* __restrict__ X,
                                          const _Float16* __restrict__ WT,
                                          int row0, int nt0, int lane,
                                          v8f acc[4]) {
  v16h a_cur = load_a_frag(X, K, row0, 0, lane);
  v16h b_cur[4];
#pragma unroll
  for (int j = 0; j < 4; ++j)
    b_cur[j] = load_bt_frag(WT, K, (nt0 + j) * 16, 0, lane);
#pragma unroll
  for (int k0 = 0; k0 < K; k0 += 32) {
    v16h a_nxt = a_cur;
    v16h b_nxt[4] = {b_cur[0], b_cur[1], b_cur[2], b_cur[3]};
    if (k0 + 32 < K) {  // folds away: K is a compile-time constant
      a_nxt = load_a_frag(X, K, row0, k0 + 32, lane);
#pragma unroll
      for (int j = 0; j < 4; ++j)
        b_nxt[j] = load_bt_frag(WT, K, (nt0 + j) * 16, k0 + 32, lane);
    }
#pragma unroll
    for (int j = 0; j < 4; ++j) acc[j] = wmma16(a_cur, b_cur[j], acc[j]);
    a_cur = a_nxt;
#pragma unroll
    for (int j = 0; j < 4; ++j) b_cur[j] = b_nxt[j];
  }
}

// ---------------- small kernels ---------------------------------------------

// Convert (K,N) f32 weights into transposed (N,K) f16 so WMMA B fragments are
// contiguous per lane.
__global__ void convert_transpose_kernel(const float* __restrict__ src,
                                         _Float16* __restrict__ dst,
                                         int K, int N) {
  int i = blockIdx.x * blockDim.x + threadIdx.x;
  if (i < K * N) {
    int k = i / N, n = i % N;
    dst[(size_t)n * K + k] = (_Float16)src[i];
  }
}

// LayerNorm; windowed==1 additionally performs roll(-3,-3) + window partition,
// producing rows in (window, token) order for the attention path.
__global__ __launch_bounds__(192) void ln_kernel(const float* __restrict__ x,
                                                 const float* __restrict__ g,
                                                 const float* __restrict__ b,
                                                 _Float16* __restrict__ out,
                                                 int windowed) {
  __shared__ float red[CC];
  int row = blockIdx.x;
  int c = threadIdx.x;
  int src;
  if (windowed) {
    int w = row / NTOK, n = row % NTOK;
    int lat = w / MLON, lon = w % MLON;
    int i = n / WSZ, j = n % WSZ;
    int hh = (lat * WSZ + i + SHIFTV) % HH;
    int ww = (lon * WSZ + j + SHIFTV) % WW_;
    src = hh * WW_ + ww;
  } else {
    src = row;
  }
  float v = x[(size_t)src * CC + c];
  red[c] = v;
  __syncthreads();
  for (int s = 96; s >= 3; s >>= 1) { if (c < s) red[c] += red[c + s]; __syncthreads(); }
  float mu = (red[0] + red[1] + red[2]) * (1.0f / CC);
  __syncthreads();
  float d = v - mu;
  red[c] = d * d;
  __syncthreads();
  for (int s = 96; s >= 3; s >>= 1) { if (c < s) red[c] += red[c + s]; __syncthreads(); }
  float var = (red[0] + red[1] + red[2]) * (1.0f / CC);
  float r = rsqrtf(var + 1e-5f);
  out[(size_t)row * CC + c] = (_Float16)(d * r * g[c] + b[c]);
}

// ---------------- GEMM kernels ----------------------------------------------
// 2 waves per block; each wave owns one 16-row M-tile and accumulates 4 N-tiles
// so every A fragment is reused by 4 WMMAs. WT is the transposed (N,K) weight.

__global__ __launch_bounds__(64) void qkv_gemm_kernel(
    const _Float16* __restrict__ X, const _Float16* __restrict__ WT,
    const float* __restrict__ bias, _Float16* __restrict__ qb,
    _Float16* __restrict__ kb, _Float16* __restrict__ vb) {
  int lane = threadIdx.x & 31, wave = threadIdx.x >> 5;
  int row0 = (blockIdx.x * 2 + wave) * 16;
  int nt0 = blockIdx.y * 4;
  v8f acc[4] = {{}, {}, {}, {}};
  gemm_core<CC>(X, WT, row0, nt0, lane, acc);
  int r0 = row0 + ((lane >> 4) << 3);
  int rw[8], rn[8];
#pragma unroll
  for (int r = 0; r < 8; ++r) { rw[r] = (r0 + r) / NTOK; rn[r] = (r0 + r) % NTOK; }
#pragma unroll
  for (int j = 0; j < 4; ++j) {
    int col = (nt0 + j) * 16 + (lane & 15);
    int sel = col / CC, rem = col % CC;
    int head = rem / HDIM, d = rem % HDIM;
    _Float16* dst = (sel == 0) ? qb : (sel == 1) ? kb : vb;
    float bv = bias[col];
#pragma unroll
    for (int r = 0; r < 8; ++r)
      dst[(((size_t)rw[r] * NHEAD + head) * NPAD + rn[r]) * HDIM + d] =
          (_Float16)(acc[j][r] + bv);
  }
}

__global__ __launch_bounds__(64) void proj_gemm_kernel(
    const _Float16* __restrict__ X, const _Float16* __restrict__ WT,
    const float* __restrict__ bias, const float* __restrict__ x_in,
    float* __restrict__ x1) {
  int lane = threadIdx.x & 31, wave = threadIdx.x >> 5;
  int row0 = (blockIdx.x * 2 + wave) * 16;
  int nt0 = blockIdx.y * 4;
  v8f acc[4] = {{}, {}, {}, {}};
  gemm_core<CC>(X, WT, row0, nt0, lane, acc);
  int r0 = row0 + ((lane >> 4) << 3);
  size_t tbase[8];
#pragma unroll
  for (int r = 0; r < 8; ++r) {
    int g = r0 + r;  // window-order row -> window reverse + roll(+3,+3)
    int w = g / NTOK, n = g % NTOK;
    int lat = w / MLON, lon = w % MLON;
    int i = n / WSZ, j = n % WSZ;
    int hh = (lat * WSZ + i + SHIFTV) % HH;
    int ww = (lon * WSZ + j + SHIFTV) % WW_;
    tbase[r] = (size_t)(hh * WW_ + ww) * CC;
  }
#pragma unroll
  for (int j = 0; j < 4; ++j) {
    int col = (nt0 + j) * 16 + (lane & 15);
    float bv = bias[col];
#pragma unroll
    for (int r = 0; r < 8; ++r) {
      size_t t = tbase[r] + col;
      x1[t] = x_in[t] + acc[j][r] + bv;  // first residual
    }
  }
}

__global__ __launch_bounds__(64) void fc1_gemm_kernel(
    const _Float16* __restrict__ X, const _Float16* __restrict__ WT,
    const float* __restrict__ bias, _Float16* __restrict__ h1) {
  int lane = threadIdx.x & 31, wave = threadIdx.x >> 5;
  int row0 = (blockIdx.x * 2 + wave) * 16;
  int nt0 = blockIdx.y * 4;
  v8f acc[4] = {{}, {}, {}, {}};
  gemm_core<CC>(X, WT, row0, nt0, lane, acc);
  int r0 = row0 + ((lane >> 4) << 3);
#pragma unroll
  for (int j = 0; j < 4; ++j) {
    int col = (nt0 + j) * 16 + (lane & 15);
    float bv = bias[col];
#pragma unroll
    for (int r = 0; r < 8; ++r) {
      float v = acc[j][r] + bv;
      float ge = 0.5f * v * (1.0f + erff(v * 0.70710678118654752f));  // exact GELU
      h1[(size_t)(r0 + r) * (4 * CC) + col] = (_Float16)ge;
    }
  }
}

__global__ __launch_bounds__(64) void fc2_gemm_kernel(
    const _Float16* __restrict__ X, const _Float16* __restrict__ WT,
    const float* __restrict__ bias, const float* __restrict__ x1,
    float* __restrict__ out) {
  int lane = threadIdx.x & 31, wave = threadIdx.x >> 5;
  int row0 = (blockIdx.x * 2 + wave) * 16;
  int nt0 = blockIdx.y * 4;
  v8f acc[4] = {{}, {}, {}, {}};
  gemm_core<4 * CC>(X, WT, row0, nt0, lane, acc);
  int r0 = row0 + ((lane >> 4) << 3);
#pragma unroll
  for (int j = 0; j < 4; ++j) {
    int col = (nt0 + j) * 16 + (lane & 15);
    float bv = bias[col];
#pragma unroll
    for (int r = 0; r < 8; ++r) {
      size_t t = (size_t)(r0 + r) * CC + col;
      out[t] = x1[t] + acc[j][r] + bv;  // second residual
    }
  }
}

// ---------------- fused attention kernel ------------------------------------

__device__ __forceinline__ int labh(int gh) {
  return gh < HH - WSZ ? 0 : (gh < HH - SHIFTV ? 1 : 2);
}
__device__ __forceinline__ int labw(int gw) {
  return gw < WW_ - WSZ ? 0 : (gw < WW_ - SHIFTV ? 1 : 2);
}

__global__ __launch_bounds__(128) void attn_kernel(
    const _Float16* __restrict__ q, const _Float16* __restrict__ k,
    const _Float16* __restrict__ v, const float* __restrict__ bias_table,
    _Float16* __restrict__ out) {
  __shared__ float    S[NPAD][NPAD];
  __shared__ _Float16 P[NPAD][64];
  __shared__ _Float16 Vt[64][HDIM];

  int wh = blockIdx.x;
  int w = wh / NHEAD, h = wh % NHEAD;
  int lat = w / MLON, lon = w % MLON;
  size_t base = ((size_t)w * NHEAD + h) * NPAD * HDIM;
  const _Float16* qb = q + base;
  const _Float16* kb = k + base;
  const _Float16* vb = v + base;
  int tid = threadIdx.x, lane = tid & 31, wave = tid >> 5;

  // V into LDS, zero-padded to 64 K-rows
  for (int idx = tid; idx < 64 * HDIM; idx += 128) {
    int m = idx >> 5, d = idx & 31;
    Vt[m][d] = (m < NTOK) ? vb[m * HDIM + d] : (_Float16)0.0f;
  }

  // S = Q @ K^T : 3x3 tiles, K-dim = head_dim = 32 (single WMMA each)
  if (wave < 3) {
    v16h a = load_a_frag(qb, HDIM, wave * 16, 0, lane);
    for (int ct = 0; ct < 3; ++ct) {
      v16h b = load_bt_frag(kb, HDIM, ct * 16, 0, lane);
      v8f c = {};
      c = wmma16(a, b, c);
      int r0 = wave * 16 + ((lane >> 4) << 3);
      int cc = ct * 16 + (lane & 15);
#pragma unroll
      for (int r = 0; r < 8; ++r) S[r0 + r][cc] = c[r];
    }
  }
  __syncthreads();

  // scale + earth-specific bias + shift mask + softmax (one thread per row)
  if (tid < NPAD) {
    int n = tid;
    if (n < NTOK) {
      int ci = n / WSZ, cj = n % WSZ;
      int regn = 3 * labh(lat * WSZ + ci) + labw(lon * WSZ + cj);
      float mx = -1e30f;
      for (int m = 0; m < NTOK; ++m) {
        int mi = m / WSZ, mj = m % WSZ;
        float bias = bias_table[(size_t)(lat * 121 + (ci - mi + 5) * 11 +
                                        (cj - mj + 5)) * NHEAD + h];
        int regm = 3 * labh(lat * WSZ + mi) + labw(lon * WSZ + mj);
        float sv = S[n][m] * SCALEV + bias + ((regm != regn) ? -100.0f : 0.0f);
        S[n][m] = sv;  // row-exclusive overwrite
        mx = fmaxf(mx, sv);
      }
      float sum = 0.0f;
      for (int m = 0; m < NTOK; ++m) {
        float e = __expf(S[n][m] - mx);
        sum += e;
        P[n][m] = (_Float16)e;
      }
      float inv = 1.0f / sum;
      for (int m = 0; m < NTOK; ++m) P[n][m] = (_Float16)((float)P[n][m] * inv);
      for (int m = NTOK; m < 64; ++m) P[n][m] = (_Float16)0.0f;
    } else {
      for (int m = 0; m < 64; ++m) P[n][m] = (_Float16)0.0f;
    }
  }
  __syncthreads();

  // O = P @ V : 3x2 tiles, K padded to 64 (2 WMMA k-steps)
  for (int t = wave; t < 6; t += 4) {
    int rt = t >> 1, ct = t & 1;
    v8f c = {};
#pragma unroll
    for (int ks = 0; ks < 2; ++ks) {
      v16h a = load_a_frag(&P[0][0], 64, rt * 16, ks * 32, lane);
      v16h b = load_b_frag(&Vt[0][0], HDIM, ks * 32, ct * 16, lane);
      c = wmma16(a, b, c);
    }
    int r0 = rt * 16 + ((lane >> 4) << 3);
    int cc = ct * 16 + (lane & 15);
#pragma unroll
    for (int r = 0; r < 8; ++r) {
      int n = r0 + r;
      if (n < NTOK)
        out[((size_t)w * NTOK + n) * CC + h * HDIM + cc] = (_Float16)c[r];
    }
  }
}

// ---------------- launch ----------------------------------------------------

extern "C" void kernel_launch(void* const* d_in, const int* in_sizes, int n_in,
                              void* d_out, int out_size, void* d_ws, size_t ws_size,
                              hipStream_t stream) {
  const float* x          = (const float*)d_in[0];
  const float* norm1_w    = (const float*)d_in[1];
  const float* norm1_b    = (const float*)d_in[2];
  const float* qkv_w      = (const float*)d_in[3];
  const float* qkv_b      = (const float*)d_in[4];
  const float* proj_w     = (const float*)d_in[5];
  const float* proj_b     = (const float*)d_in[6];
  const float* bias_table = (const float*)d_in[7];
  const float* norm2_w    = (const float*)d_in[8];
  const float* norm2_b    = (const float*)d_in[9];
  const float* fc1_w      = (const float*)d_in[10];
  const float* fc1_b      = (const float*)d_in[11];
  const float* fc2_w      = (const float*)d_in[12];
  const float* fc2_b      = (const float*)d_in[13];
  float* out = (float*)d_out;

  char* ws = (char*)d_ws;
  size_t off = 0;
  auto alloc = [&](size_t bytes) {
    char* p = ws + off;
    off = (off + bytes + 255) & ~(size_t)255;
    return p;
  };

  // Transposed (N,K) f16 weights
  _Float16* wqkvT = (_Float16*)alloc((size_t)3 * CC * CC * 2);
  _Float16* wprojT = (_Float16*)alloc((size_t)CC * CC * 2);
  _Float16* wfc1T = (_Float16*)alloc((size_t)4 * CC * CC * 2);
  _Float16* wfc2T = (_Float16*)alloc((size_t)CC * 4 * CC * 2);
  _Float16* xw = (_Float16*)alloc((size_t)TOKENS * CC * 2);
  size_t qkvsz = (size_t)NWIN * NHEAD * NPAD * HDIM * 2;
  _Float16* qbuf = (_Float16*)alloc(qkvsz);
  _Float16* kbuf = (_Float16*)alloc(qkvsz);
  _Float16* vbuf = (_Float16*)alloc(qkvsz);
  _Float16* ao = (_Float16*)alloc((size_t)TOKENS * CC * 2);
  float* x1 = (float*)alloc((size_t)TOKENS * CC * 4);
  _Float16* mbuf = (_Float16*)alloc((size_t)TOKENS * CC * 2);
  _Float16* h1 = (_Float16*)alloc((size_t)TOKENS * 4 * CC * 2);

  // 1) weights -> transposed f16
  convert_transpose_kernel<<<(CC * 3 * CC + 255) / 256, 256, 0, stream>>>(
      qkv_w, wqkvT, CC, 3 * CC);
  convert_transpose_kernel<<<(CC * CC + 255) / 256, 256, 0, stream>>>(
      proj_w, wprojT, CC, CC);
  convert_transpose_kernel<<<(CC * 4 * CC + 255) / 256, 256, 0, stream>>>(
      fc1_w, wfc1T, CC, 4 * CC);
  convert_transpose_kernel<<<(4 * CC * CC + 255) / 256, 256, 0, stream>>>(
      fc2_w, wfc2T, 4 * CC, CC);

  // 2) LN1 + roll + window partition
  ln_kernel<<<TOKENS, CC, 0, stream>>>(x, norm1_w, norm1_b, xw, 1);

  // 3) QKV GEMM (64800x192 @ 192x576)
  qkv_gemm_kernel<<<dim3(TOKENS / 32, (3 * CC / 16) / 4), 64, 0, stream>>>(
      xw, wqkvT, qkv_b, qbuf, kbuf, vbuf);

  // 4) windowed attention (one block per window*head)
  attn_kernel<<<NWIN * NHEAD, 128, 0, stream>>>(qbuf, kbuf, vbuf, bias_table, ao);

  // 5) proj GEMM + window reverse + roll back + residual
  proj_gemm_kernel<<<dim3(TOKENS / 32, (CC / 16) / 4), 64, 0, stream>>>(
      ao, wprojT, proj_b, x, x1);

  // 6) LN2
  ln_kernel<<<TOKENS, CC, 0, stream>>>(x1, norm2_w, norm2_b, mbuf, 0);

  // 7) MLP
  fc1_gemm_kernel<<<dim3(TOKENS / 32, (4 * CC / 16) / 4), 64, 0, stream>>>(
      mbuf, wfc1T, fc1_b, h1);
  fc2_gemm_kernel<<<dim3(TOKENS / 32, (CC / 16) / 4), 64, 0, stream>>>(
      h1, wfc2T, fc2_b, x1, out);

  (void)in_sizes; (void)n_in; (void)out_size; (void)ws_size;
}
